// NeuralKB_37701222924639
// MI455X (gfx1250) — matrix operand order repaired and checked by reading, not compiled, not run on hardware
//
#include <hip/hip_runtime.h>
#include <hip/hip_bf16.h>
#include <stdint.h>

typedef __bf16 bf16_t;
typedef __attribute__((ext_vector_type(16))) __bf16 v16bf;
typedef __attribute__((ext_vector_type(8)))  __bf16 v8bf;
typedef __attribute__((ext_vector_type(8)))  float  v8f;

#define DSUB 128          // per-part embedding dim
#define KDIM 384          // 3*DSUB
#define TN   64           // fact columns per workgroup (tail clamped)
#define NSUB 4            // TN/16 accumulators per wave
#define LSTR 392          // padded LDS k-stride in bf16 elems (784B -> conflict-free b128)

static __device__ __forceinline__ v16bf cat8(v8bf a, v8bf b) {
  return __builtin_shufflevector(a, b, 0,1,2,3,4,5,6,7,8,9,10,11,12,13,14,15);
}

static __device__ __forceinline__ v8f wmma_bf16(v16bf a, v16bf b, v8f c) {
  return __builtin_amdgcn_wmma_f32_16x16x32_bf16(false, a, false, b, (short)0, c, false, false);
}

// order-preserving map float -> uint (monotone for all finite values)
static __device__ __forceinline__ unsigned int ordkey(float v) {
  unsigned int u = __float_as_uint(v);
  return (u & 0x80000000u) ? ~u : (u | 0x80000000u);
}

// min-reduce a u64 key across groups of 16 lanes
static __device__ __forceinline__ unsigned long long minred16(unsigned long long k) {
  #pragma unroll
  for (int m = 1; m <= 8; m <<= 1) {
    unsigned int hi = (unsigned int)(k >> 32);
    unsigned int lo = (unsigned int)(k & 0xFFFFFFFFull);
    unsigned int ohi = __shfl_xor(hi, m, 32);
    unsigned int olo = __shfl_xor(lo, m, 32);
    unsigned long long o = ((unsigned long long)ohi << 32) | (unsigned long long)olo;
    if (o < k) k = o;
  }
  return k;
}

// element k (0..383) of the concatenated [rel|arg1|arg2] row
static __device__ __forceinline__ float cat_read(const float* __restrict__ p0,
                                                 const float* __restrict__ p1,
                                                 const float* __restrict__ p2,
                                                 int row, int k) {
  const float* p = (k < DSUB)     ? (p0 + (size_t)row * DSUB + k)
                 : (k < 2*DSUB)   ? (p1 + (size_t)row * DSUB + (k - DSUB))
                                  : (p2 + (size_t)row * DSUB + (k - 2*DSUB));
  return *p;
}

// A fragment (16x32 bf16): lanes 0-15 -> K[ks+0..7],[ks+16..23]; lanes 16-31 -> +8
static __device__ __forceinline__ v16bf loadA(const bf16_t* __restrict__ row, int ks, int lhalf) {
  return cat8(*(const v8bf*)(row + ks + lhalf * 8),
              *(const v8bf*)(row + ks + 16 + lhalf * 8));
}

// ---------------------------------------------------------------------------
// Kernel 1: pack batch rows to bf16 hi/lo (row-major [B][384]) + init keys
// ---------------------------------------------------------------------------
__global__ __launch_bounds__(256) void pack_batch(
    const float* __restrict__ rel, const float* __restrict__ a1, const float* __restrict__ a2,
    bf16_t* __restrict__ bhi, bf16_t* __restrict__ blo,
    unsigned long long* __restrict__ keys, int B) {
  int i = blockIdx.x * 256 + threadIdx.x;
  if (i < B) keys[i] = 0xFFFFFFFFFFFFFFFFull;
  if (i >= B * KDIM) return;
  int b = i / KDIM;
  int k = i - b * KDIM;
  float x = cat_read(rel, a1, a2, b, k);
  bf16_t h = (bf16_t)x;
  bf16_t l = (bf16_t)(x - (float)h);
  bhi[i] = h;
  blo[i] = l;
}

// ---------------------------------------------------------------------------
// Kernel 2: per-fact squared norms (one wave32 per fact row)
// ---------------------------------------------------------------------------
__global__ __launch_bounds__(256) void fact_norms(
    const float* __restrict__ frel, const float* __restrict__ f1, const float* __restrict__ f2,
    float* __restrict__ fnorm, int N) {
  int w    = (blockIdx.x * 256 + threadIdx.x) >> 5;
  int lane = threadIdx.x & 31;
  if (w >= N) return;
  float s = 0.f;
  #pragma unroll
  for (int k = lane; k < KDIM; k += 32) {
    float x = cat_read(frel, f1, f2, w, k);
    s += x * x;
  }
  #pragma unroll
  for (int m = 16; m >= 1; m >>= 1) s += __shfl_xor(s, m, 32);
  if (lane == 0) fnorm[w] = s;
}

// ---------------------------------------------------------------------------
// Kernel 3: split-bf16 WMMA GEMM + running argmin of (||f||^2 - 2 b.f)
//   grid.x = ceil(N/64), 256 threads (8 waves). Fact slab fp32->bf16 in LDS.
// ---------------------------------------------------------------------------
__global__ __launch_bounds__(256) void gemm_min(
    const float* __restrict__ frel, const float* __restrict__ f1, const float* __restrict__ f2,
    const bf16_t* __restrict__ bhi, const bf16_t* __restrict__ blo,
    const float* __restrict__ fnorm, unsigned long long* __restrict__ keys,
    int B, int N) {
  __shared__ __attribute__((aligned(16))) bf16_t shi[TN * LSTR];
  __shared__ __attribute__((aligned(16))) bf16_t slo[TN * LSTR];
  __shared__ unsigned long long skey[1024];

  const int tid   = threadIdx.x;
  const int nBase = blockIdx.x * TN;

  for (int i = tid; i < B; i += 256) skey[i] = 0xFFFFFFFFFFFFFFFFull;

  // cooperative slab fill: TN cols x 384 K, fp32 -> (hi, lo) bf16; tail clamped
  for (int i = tid; i < TN * KDIM; i += 256) {
    int col = i / KDIM;
    int k   = i - col * KDIM;
    int n   = nBase + col; if (n > N - 1) n = N - 1;
    float x = cat_read(frel, f1, f2, n, k);
    bf16_t h = (bf16_t)x;
    bf16_t l = (bf16_t)(x - (float)h);
    shi[col * LSTR + k] = h;
    slo[col * LSTR + k] = l;
  }

  const int wave  = tid >> 5;
  const int lane  = tid & 31;
  const int lhalf = lane >> 4;   // 0 or 1
  const int l16   = lane & 15;

  int   nidx[NSUB];
  float fnv[NSUB];
  #pragma unroll
  for (int s = 0; s < NSUB; s++) {
    int n = nBase + s * 16 + l16; if (n > N - 1) n = N - 1;
    nidx[s] = n;
    fnv[s]  = fnorm[n];
  }

  const int iters = (B >> 4) >> 3;   // mTiles / 8 waves (B=1024 -> 8)
  for (int it = 0; it < iters; ++it) {
    const int mt    = it * 8 + wave;
    const int mBase = mt << 4;
    const bf16_t* ahRow = bhi + (size_t)(mBase + l16) * KDIM;
    const bf16_t* alRow = blo + (size_t)(mBase + l16) * KDIM;

    // prefetch first A fragments before the barrier
    v16bf ah_c = loadA(ahRow, 0, lhalf);
    v16bf al_c = loadA(alRow, 0, lhalf);

    __syncthreads();   // also fences LDS: keeps B fragments from being hoisted to regs

    v8f acc[NSUB];
    #pragma unroll
    for (int s = 0; s < NSUB; s++) acc[s] = (v8f){};

    #pragma unroll
    for (int ks = 0; ks < KDIM; ks += 32) {
      // double-buffer: issue next A loads while 12 WMMAs run on current
      v16bf ah_n = ah_c, al_n = al_c;
      if (ks + 32 < KDIM) {
        ah_n = loadA(ahRow, ks + 32, lhalf);
        al_n = loadA(alRow, ks + 32, lhalf);
      }
      #pragma unroll
      for (int s = 0; s < NSUB; s++) {
        const bf16_t* bh = shi + (s * 16 + l16) * LSTR + ks + lhalf * 8;
        const bf16_t* bl = slo + (s * 16 + l16) * LSTR + ks + lhalf * 8;
        v16bf bhv = cat8(*(const v8bf*)bh, *(const v8bf*)(bh + 16));
        v16bf blv = cat8(*(const v8bf*)bl, *(const v8bf*)(bl + 16));
        acc[s] = wmma_bf16(ah_c, bhv, acc[s]);   // hi*hi
        acc[s] = wmma_bf16(ah_c, blv, acc[s]);   // hi*lo
        acc[s] = wmma_bf16(al_c, bhv, acc[s]);   // lo*hi
      }
      ah_c = ah_n; al_c = al_n;
    }

    // C layout: VGPR j, lane L -> row mBase + j + 8*(L>>4), col = L&15 of each subtile.
    // All subtiles of a given j share the row -> min-select in-lane first.
    #pragma unroll
    for (int j = 0; j < 8; j++) {
      int   row  = mBase + j + (lhalf << 3);
      float vmin = fnv[0] - 2.0f * acc[0][j];
      int   nmin = nidx[0];
      #pragma unroll
      for (int s = 1; s < NSUB; s++) {
        float v = fnv[s] - 2.0f * acc[s][j];
        if (v < vmin) { vmin = v; nmin = nidx[s]; }
      }
      unsigned long long key =
          ((unsigned long long)ordkey(vmin) << 32) | (unsigned int)nmin;
      key = minred16(key);
      if (l16 == 0) atomicMin(&skey[row], key);
    }
  }
  __syncthreads();
  for (int i = tid; i < B; i += 256) atomicMin(&keys[i], skey[i]);
}

// ---------------------------------------------------------------------------
// Kernel 4: exact fp32 rescore of the winner (one wave32 per batch row)
// ---------------------------------------------------------------------------
__global__ __launch_bounds__(256) void rescore(
    const float* __restrict__ rel, const float* __restrict__ a1, const float* __restrict__ a2,
    const float* __restrict__ frel, const float* __restrict__ f1, const float* __restrict__ f2,
    const unsigned long long* __restrict__ keys, float* __restrict__ out, int B) {
  int b    = (blockIdx.x * 256 + threadIdx.x) >> 5;
  int lane = threadIdx.x & 31;
  if (b >= B) return;
  int n = (int)(unsigned int)(keys[b] & 0xFFFFFFFFull);
  float s = 0.f;
  #pragma unroll
  for (int k = lane; k < KDIM; k += 32) {
    float e = cat_read(rel, a1, a2, b, k) - cat_read(frel, f1, f2, n, k);
    s += e * e;
  }
  #pragma unroll
  for (int m = 16; m >= 1; m >>= 1) s += __shfl_xor(s, m, 32);
  if (lane == 0) out[b] = expf(-0.5f * s);
}

// ---------------------------------------------------------------------------
extern "C" void kernel_launch(void* const* d_in, const int* in_sizes, int n_in,
                              void* d_out, int out_size, void* d_ws, size_t ws_size,
                              hipStream_t stream) {
  const float* rel  = (const float*)d_in[0];
  const float* a1   = (const float*)d_in[1];
  const float* a2   = (const float*)d_in[2];
  const float* frel = (const float*)d_in[3];
  const float* f1   = (const float*)d_in[4];
  const float* f2   = (const float*)d_in[5];
  float* out = (float*)d_out;

  const int B = in_sizes[0] / DSUB;   // 1024
  const int N = in_sizes[3] / DSUB;   // 100000

  char* w = (char*)d_ws;
  unsigned long long* keys = (unsigned long long*)w;
  size_t off = (size_t)B * 8;                               // keys
  float* fnorm = (float*)(w + off); off += (size_t)N * 4;   // fact norms
  off = (off + 15) & ~(size_t)15;
  bf16_t* bhi = (bf16_t*)(w + off); off += (size_t)B * KDIM * 2;
  bf16_t* blo = (bf16_t*)(w + off);

  pack_batch<<<(B * KDIM + 255) / 256, 256, 0, stream>>>(rel, a1, a2, bhi, blo, keys, B);
  fact_norms<<<(N + 7) / 8, 256, 0, stream>>>(frel, f1, f2, fnorm, N);
  gemm_min<<<(N + TN - 1) / TN, 256, 0, stream>>>(frel, f1, f2, bhi, blo, fnorm, keys, B, N);
  rescore<<<(B + 7) / 8, 256, 0, stream>>>(rel, a1, a2, frel, f1, f2, keys, out, B);
}
